// STMA_73409581023888
// MI455X (gfx1250) — matrix operand order repaired and checked by reading, not compiled, not run on hardware
//
// STMA forward for MI455X (gfx1250): bf16 WMMA GEMMs + wave32 fp32 epilogues.
// ~125 GFLOP total; memory floor is a few us at 23.3 TB/s, so all dense matmuls
// run on v_wmma_f32_16x16x32_bf16 (fp32 accum). All GEMM operands are arranged
// so A is row-major-contiguous and B is pre-transposed ([N,K]) -> each WMMA
// fragment is fed by global_load_b128 vector loads. The k-loop is software-
// pipelined with explicit ping-pong buffers (2x unroll) so next-step loads
// overlap current-step WMMAs with no register-copy traffic between buffers.
#include <hip/hip_runtime.h>
#include <hip/hip_bf16.h>
#include <math.h>

typedef __bf16 bf16_t;
typedef __attribute__((ext_vector_type(16))) __bf16 v16bf;
typedef __attribute__((ext_vector_type(8)))  __bf16 v8bf;
typedef __attribute__((ext_vector_type(8)))  float  v8f;

// ---------------- problem dims ----------------
#define cB    64
#define cT    256
#define cNt   77
#define cH    8
#define cL    64
#define cD    512
#define cPt   256
#define cE    8
#define cTEMB 2048
#define NTOK_T (cB * cNt)        // 4928  (= 308*16)
#define NTOK_M (cB * cT * cH)    // 131072
#define cNN   (cNt + cT)         // 333
#define cNPAD 352                // padded to multiple of 32 (352*2B = 16B-aligned rows)
#define NEGV  (-1.0e6f)

// ---------------- small device helpers ----------------
__device__ __forceinline__ float wave_sum(float v) {
#pragma unroll
  for (int o = 16; o > 0; o >>= 1) v += __shfl_xor(v, o, 32);
  return v;
}
__device__ __forceinline__ float wave_max(float v) {
#pragma unroll
  for (int o = 16; o > 0; o >>= 1) v = fmaxf(v, __shfl_xor(v, o, 32));
  return v;
}
__device__ __forceinline__ bf16_t to_bf16(float x) {
  union { float f; unsigned u; } v; v.f = x;
  unsigned r = v.u + 0x7fffu + ((v.u >> 16) & 1u);   // RNE
  unsigned short h = (unsigned short)(r >> 16);
  return __builtin_bit_cast(bf16_t, h);
}
__device__ __forceinline__ float gelu_f(float x) {
  return 0.5f * x * (1.0f + erff(x * 0.70710678118654752f));
}
__device__ __forceinline__ float silu_f(float x) {
  return x / (1.0f + expf(-x));
}
__device__ __forceinline__ v16bf cat16(v8bf lo, v8bf hi) {
  return __builtin_shufflevector(lo, hi, 0, 1, 2, 3, 4, 5, 6, 7,
                                 8, 9, 10, 11, 12, 13, 14, 15);
}
__device__ __forceinline__ v16bf ld_frag32(const bf16_t* p, int stride2nd) {
  // 32 consecutive-bf16 split as two 16B vector loads at p and p+stride2nd
  return cat16(*(const v8bf*)p, *(const v8bf*)(p + stride2nd));
}

// ---------------- elementwise converters ----------------
__global__ void k_cvt_bf16(const float* __restrict__ in, bf16_t* __restrict__ out, long n) {
  long i = (long)blockIdx.x * blockDim.x + threadIdx.x;
  long s = (long)gridDim.x * blockDim.x;
  for (; i < n; i += s) out[i] = to_bf16(in[i]);
}
// batched transpose+convert: in = batch of [R,C] f32, out = batch of [C,R] bf16
__global__ void k_cvt_t_bf16(const float* __restrict__ in, bf16_t* __restrict__ out,
                             int R, int C, long n) {
  long i = (long)blockIdx.x * blockDim.x + threadIdx.x;
  long s = (long)gridDim.x * blockDim.x;
  const long RC = (long)R * C;
  for (; i < n; i += s) {
    const long e = i / RC;
    const long rem = i - e * RC;
    const int r = (int)(rem / C);
    const int c = (int)(rem - (long)r * C);
    out[e * RC + (long)c * R + r] = to_bf16(in[i]);
  }
}
__global__ void k_gelu_cvt_bf16(const float* __restrict__ in, bf16_t* __restrict__ out, long n) {
  long i = (long)blockIdx.x * blockDim.x + threadIdx.x;
  long s = (long)gridDim.x * blockDim.x;
  for (; i < n; i += s) out[i] = to_bf16(gelu_f(in[i]));
}
__global__ void k_silu_cvt_bf16(const float* __restrict__ in, bf16_t* __restrict__ out, long n) {
  long i = (long)blockIdx.x * blockDim.x + threadIdx.x;
  long s = (long)gridDim.x * blockDim.x;
  for (; i < n; i += s) out[i] = to_bf16(silu_f(in[i]));
}

// ---------------- generic batched WMMA GEMM ----------------
// C[M,N] = A[M,K] (row-major, lda) x BT[N,K]^T (row-major, ldbt), batched.
// One wave computes a 16(M) x 64(N) strip; A fragment reused across 4 WMMAs.
// All fragment feeds are 16B-aligned vector loads (lda/ldbt multiples of 8).
enum { EPI_GELU_BF16 = 0, EPI_SCALED_ADD = 1, EPI_F32 = 2, EPI_RESID = 3, EPI_BF16_T = 4 };

#define LOAD_SET(av, bv, kk)                                     \
  do {                                                           \
    av = ld_frag32(arow + (kk), 16);                             \
    _Pragma("unroll")                                            \
    for (int t = 0; t < 4; ++t) bv[t] = ld_frag32(brow[t] + (kk), 16); \
  } while (0)

#define WMMA_SET(av, bv)                                         \
  do {                                                           \
    _Pragma("unroll")                                            \
    for (int t = 0; t < 4; ++t)                                  \
      acc[t] = __builtin_amdgcn_wmma_f32_16x16x32_bf16(          \
          false, av, false, bv[t], (short)0, acc[t], false, false); \
  } while (0)

template <int EPI>
__global__ __launch_bounds__(256)
void k_wmma_gemm(const bf16_t* __restrict__ A, long lda, long a_bs,
                 const bf16_t* __restrict__ BT, long ldbt, long b_bs,
                 const float* __restrict__ bias,
                 const float* __restrict__ scale, int scale_stride, int scale_off,
                 const float* __restrict__ resid,
                 void* __restrict__ Out, long o_rs, long o_bs,
                 int M, int N, int K)
{
  const int wid  = blockIdx.x * (blockDim.x >> 5) + (threadIdx.x >> 5);
  const int lane = threadIdx.x & 31;
  const int nStrips = N >> 6;
  const int total = (M >> 4) * nStrips;
  if (wid >= total) return;                 // wave-uniform: EXEC stays all-ones
  const int mt = wid / nStrips;
  const int ns = wid - mt * nStrips;
  const int row0 = mt << 4, col0 = ns << 6;
  const long batch = blockIdx.y;
  A  += batch * a_bs;
  BT += batch * b_bs;

  v8f acc[4];
#pragma unroll
  for (int t = 0; t < 4; ++t)
#pragma unroll
    for (int r = 0; r < 8; ++r) acc[t][r] = 0.0f;

  const int la = lane & 15;
  // Per-lane base pointers (ISA 16-bit fragment layouts):
  // A 16x32: lanes<16 hold K {0..7,16..23}, lanes>=16 hold K {8..15,24..31}.
  const bf16_t* arow = A + (long)(row0 + la) * lda + ((lane & 16) ? 8 : 0);
  // B 32x16: lanes<16 hold K {0..7,16..23}, lanes>=16 hold K {8..15,24..31} of col (lane&15).
  const bf16_t* brow[4];
#pragma unroll
  for (int t = 0; t < 4; ++t)
    brow[t] = BT + (long)(col0 + (t << 4) + la) * ldbt + ((lane & 16) ? 8 : 0);

  // Ping-pong software pipeline (no buffer-to-buffer copies):
  // invariant at loop head: buf0 holds k-step (k-32), not yet consumed.
  v16bf a0, a1, b0[4], b1[4];
  LOAD_SET(a0, b0, 0);
  int k = 32;
  for (; k + 32 < K; k += 64) {
    __builtin_prefetch(arow + k + 64, 0, 1);
    LOAD_SET(a1, b1, k);        // step k
    WMMA_SET(a0, b0);           // consume step k-32
    LOAD_SET(a0, b0, k + 32);   // step k+32
    WMMA_SET(a1, b1);           // consume step k
  }
  if (k < K) {                  // one unloaded step remains
    LOAD_SET(a1, b1, k);
    WMMA_SET(a0, b0);
    WMMA_SET(a1, b1);
  } else {                      // everything loaded; one buffered step remains
    WMMA_SET(a0, b0);
  }

  // C/D layout: lane holds col (lane&15); rows rbase..rbase+7
  const int rbase = row0 + ((lane & 16) ? 8 : 0);
#pragma unroll
  for (int t = 0; t < 4; ++t) {
    const int c = col0 + (t << 4) + la;
#pragma unroll
    for (int r = 0; r < 8; ++r) {
      const int rr = rbase + r;
      float v = acc[t][r];
      if (bias) v += bias[c];
      if (EPI == EPI_GELU_BF16) {
        ((bf16_t*)Out)[batch * o_bs + (long)rr * o_rs + c] = to_bf16(gelu_f(v));
      } else if (EPI == EPI_SCALED_ADD) {
        const float w = scale[(long)rr * scale_stride + scale_off];
        ((float*)Out)[batch * o_bs + (long)rr * o_rs + c] += w * v;
      } else if (EPI == EPI_F32) {
        ((float*)Out)[batch * o_bs + (long)rr * o_rs + c] = v;
      } else if (EPI == EPI_RESID) {
        const long oidx = batch * o_bs + (long)rr * o_rs + c;
        ((float*)Out)[oidx] = resid[oidx] + v;
      } else { // EPI_BF16_T: store D transposed (row rr -> col, col c -> row)
        ((bf16_t*)Out)[batch * o_bs + (long)c * o_rs + rr] = to_bf16(v);
      }
    }
  }
}

// ---------------- MoE prep: LN + pos + top-2 gate ----------------
__device__ __forceinline__ void top2_store(const float* lg, float* gates_row) {
  float mx = lg[0];
#pragma unroll
  for (int e = 1; e < cE; ++e) mx = fmaxf(mx, lg[e]);
  float p[cE]; float se = 0.f;
#pragma unroll
  for (int e = 0; e < cE; ++e) { p[e] = expf(lg[e] - mx); se += p[e]; }
#pragma unroll
  for (int e = 0; e < cE; ++e) p[e] /= se;
  int i1 = 0;
#pragma unroll
  for (int e = 1; e < cE; ++e) if (p[e] > p[i1]) i1 = e;
  int i2 = (i1 == 0) ? 1 : 0;
#pragma unroll
  for (int e = 0; e < cE; ++e) if (e != i1 && p[e] > p[i2]) i2 = e;
  const float s2 = p[i1] + p[i2];
#pragma unroll
  for (int e = 0; e < cE; ++e)
    gates_row[e] = (e == i1 || e == i2) ? (p[e] / s2) : 0.f;
}

// text tokens: one wave per token (Pt=256 -> 8 elems/lane)
__global__ void k_text_prep(const float* __restrict__ xf,
                            const float* __restrict__ g, const float* __restrict__ b,
                            const float* __restrict__ tpos, const float* __restrict__ gatew,
                            bf16_t* __restrict__ aout, float* __restrict__ gates)
{
  const int wid  = blockIdx.x * (blockDim.x >> 5) + (threadIdx.x >> 5);
  const int lane = threadIdx.x & 31;
  if (wid >= NTOK_T) return;
  const int n = wid % cNt;
  const float* row = xf + (long)wid * cPt;
  float v[8]; float s = 0.f;
#pragma unroll
  for (int j = 0; j < 8; ++j) { v[j] = row[lane + 32 * j]; s += v[j]; }
  const float mean = wave_sum(s) * (1.0f / cPt);
  float var = 0.f;
#pragma unroll
  for (int j = 0; j < 8; ++j) { const float d = v[j] - mean; var += d * d; }
  var = wave_sum(var) * (1.0f / cPt);
  const float inv = rsqrtf(var + 1e-5f);
  float lg[cE];
#pragma unroll
  for (int e = 0; e < cE; ++e) lg[e] = 0.f;
#pragma unroll
  for (int j = 0; j < 8; ++j) {
    const int d = lane + 32 * j;
    const float xn = (v[j] - mean) * inv * g[d] + b[d] + tpos[(long)n * cPt + d];
    aout[(long)wid * cPt + d] = to_bf16(xn);
#pragma unroll
    for (int e = 0; e < cE; ++e) lg[e] += xn * gatew[(long)d * cE + e];
  }
#pragma unroll
  for (int e = 0; e < cE; ++e) lg[e] = wave_sum(lg[e]);
  if (lane == 0) top2_store(lg, gates + (long)wid * cE);
}

// motion tokens: one wave per token (L=64 -> 2 elems/lane)
__global__ void k_motion_prep(const float* __restrict__ x,
                              const float* __restrict__ g, const float* __restrict__ b,
                              const float* __restrict__ mpos, const float* __restrict__ gatew,
                              bf16_t* __restrict__ aout, float* __restrict__ gates)
{
  const int wid  = blockIdx.x * (blockDim.x >> 5) + (threadIdx.x >> 5);
  const int lane = threadIdx.x & 31;
  if (wid >= NTOK_M) return;
  const int h = wid % cH;
  const int t = (wid / cH) % cT;
  const float* row = x + (long)wid * cL;
  const float v0 = row[lane], v1 = row[lane + 32];
  const float mean = wave_sum(v0 + v1) * (1.0f / cL);
  const float d0 = v0 - mean, d1 = v1 - mean;
  const float var = wave_sum(d0 * d0 + d1 * d1) * (1.0f / cL);
  const float inv = rsqrtf(var + 1e-5f);
  const float* pos = mpos + ((long)t * cH + h) * cL;
  const float xn0 = d0 * inv * g[lane]      + b[lane]      + pos[lane];
  const float xn1 = d1 * inv * g[lane + 32] + b[lane + 32] + pos[lane + 32];
  aout[(long)wid * cL + lane]      = to_bf16(xn0);
  aout[(long)wid * cL + lane + 32] = to_bf16(xn1);
  float lg[cE];
#pragma unroll
  for (int e = 0; e < cE; ++e)
    lg[e] = wave_sum(xn0 * gatew[(long)lane * cE + e] + xn1 * gatew[(long)(lane + 32) * cE + e]);
  if (lane == 0) top2_store(lg, gates + (long)wid * cE);
}

// ---------------- body-part mixing ----------------
__global__ void k_bodymix(const float* __restrict__ motion_feat,
                          const float* __restrict__ body_w,
                          float* __restrict__ body_feat)
{
  const long idx = (long)blockIdx.x * blockDim.x + threadIdx.x;
  if (idx >= (long)cB * cT * cH * cL) return;
  const int d = (int)(idx & 63);
  const long r = idx >> 6;
  const int h = (int)(r & 7);
  const long bt = r >> 3;              // b*T + t
  float w[cH]; float mx = body_w[h * cH];
#pragma unroll
  for (int l = 1; l < cH; ++l) mx = fmaxf(mx, body_w[h * cH + l]);
  float se = 0.f;
#pragma unroll
  for (int l = 0; l < cH; ++l) { w[l] = expf(body_w[h * cH + l] - mx); se += w[l]; }
  float acc = 0.f;
#pragma unroll
  for (int l = 0; l < cH; ++l)
    acc += (w[l] / se) * motion_feat[(bt * cH + l) * 256 + d];
  body_feat[idx] = acc;
}

// ---------------- attention key/value/query builders ----------------
__device__ __forceinline__ float key_logit(const float* tf, const float* mf,
                                           const float* sm, float tcneg,
                                           int b, int h, int d, int n) {
  if (n < cNt) return tf[((long)b * cNt + n) * 128 + d] + tcneg;
  const int t = n - cNt;
  const float m = sm[(long)b * cT + t];
  return mf[(((long)b * cT + t) * cH + h) * 256 + 64 + d] + (1.0f - m) * NEGV;
}

// softmax over N per (b,h,d); output keyPT [b,h][d][NPAD] (contiguous n stores)
__global__ void k_build_key(const float* __restrict__ tf, const float* __restrict__ mf,
                            const float* __restrict__ sm, const int* __restrict__ cond,
                            bf16_t* __restrict__ keyPT)
{
  const int wid  = blockIdx.x * (blockDim.x >> 5) + (threadIdx.x >> 5);
  const int lane = threadIdx.x & 31;
  if (wid >= cB * cH * 2) return;
  const int dg = wid & 1;
  const int bh = wid >> 1;
  const int h = bh & 7, b = bh >> 3;
  const int d = dg * 32 + lane;
  const float tcneg = ((cond[b] % 10) > 0) ? 0.f : NEGV;
  float m = -3.0e38f, s = 0.f;
  for (int n = 0; n < cNN; ++n) {
    const float lg = key_logit(tf, mf, sm, tcneg, b, h, d, n);
    const float mn = fmaxf(m, lg);
    s = s * expf(m - mn) + expf(lg - mn);
    m = mn;
  }
  const float inv = 1.0f / s;
  bf16_t* out = keyPT + ((long)bh * 64 + d) * cNPAD;
  for (int n = 0; n < cNPAD; ++n) {
    float v = 0.f;
    if (n < cNN) v = expf(key_logit(tf, mf, sm, tcneg, b, h, d, n) - m) * inv;
    out[n] = to_bf16(v);
  }
}

// valPT [b,h][l][NPAD]
__global__ void k_build_value(const float* __restrict__ tf, const float* __restrict__ mf,
                              const float* __restrict__ sm, const int* __restrict__ cond,
                              bf16_t* __restrict__ valPT)
{
  const long idx = (long)blockIdx.x * blockDim.x + threadIdx.x;
  if (idx >= (long)cB * cH * 64 * cNPAD) return;
  const int n = (int)(idx % cNPAD);
  const long rest = idx / cNPAD;
  const int l = (int)(rest & 63);
  const int bh = (int)(rest >> 6);
  const int h = bh & 7, b = bh >> 3;
  float v = 0.f;
  if (n < cNt) {
    const float tc = ((cond[b] % 10) > 0) ? 1.f : 0.f;
    v = tf[((long)b * cNt + n) * 128 + 64 + l] * tc;
  } else if (n < cNN) {
    const int t = n - cNt;
    v = mf[(((long)b * cT + t) * cH + h) * 256 + 128 + l] * sm[(long)b * cT + t];
  }
  valPT[idx] = to_bf16(v);
}

// query softmax over L=64, output layout [B,H,T,64]
__global__ void k_query(const float* __restrict__ mf, bf16_t* __restrict__ qbf)
{
  const int wid  = blockIdx.x * (blockDim.x >> 5) + (threadIdx.x >> 5);
  const int lane = threadIdx.x & 31;
  if (wid >= cB * cH * cT) return;
  const int t = wid % cT;
  const int bh = wid / cT;
  const int h = bh % cH, b = bh / cH;
  const float* src = mf + (((long)b * cT + t) * cH + h) * 256 + 192;
  const float v0 = src[lane], v1 = src[lane + 32];
  const float mx = wave_max(fmaxf(v0, v1));
  const float e0 = expf(v0 - mx), e1 = expf(v1 - mx);
  const float inv = 1.0f / wave_sum(e0 + e1);
  bf16_t* dst = qbf + (long)wid * 64;
  dst[lane]      = to_bf16(e0 * inv);
  dst[lane + 32] = to_bf16(e1 * inv);
}

// ---------------- stylization: LN + modulate + SiLU -> bf16 ----------------
__global__ void k_styl(const float* __restrict__ body_feat, const float* __restrict__ y_attn,
                       const float* __restrict__ eo,
                       const float* __restrict__ ng, const float* __restrict__ nb,
                       bf16_t* __restrict__ hsil)
{
  const int wid  = blockIdx.x * (blockDim.x >> 5) + (threadIdx.x >> 5);
  const int lane = threadIdx.x & 31;
  if (wid >= cB * cT) return;
  const int b = wid / cT, t = wid % cT;
  float v[16]; float s = 0.f;
#pragma unroll
  for (int j = 0; j < 16; ++j) {
    const int d = lane + 32 * j;
    const int h = d >> 6, l = d & 63;
    const float sv = body_feat[(long)wid * cD + d] +
                     y_attn[(((long)b * cH + h) * cT + t) * 64 + l];
    v[j] = sv; s += sv;
  }
  const float mean = wave_sum(s) * (1.0f / cD);
  float var = 0.f;
#pragma unroll
  for (int j = 0; j < 16; ++j) { const float d0 = v[j] - mean; var += d0 * d0; }
  var = wave_sum(var) * (1.0f / cD);
  const float inv = rsqrtf(var + 1e-5f);
#pragma unroll
  for (int j = 0; j < 16; ++j) {
    const int d = lane + 32 * j;
    const float hn = (v[j] - mean) * inv * ng[d] + nb[d];
    const float sc = eo[(long)b * 1024 + d];
    const float sh = eo[(long)b * 1024 + 512 + d];
    const float hm = hn * (1.0f + sc) + sh;
    hsil[(long)wid * cD + d] = to_bf16(silu_f(hm));
  }
}

// ---------------- host-side launch helpers ----------------
template <int EPI>
static void gemm(hipStream_t st,
                 const bf16_t* A, long lda, long a_bs,
                 const bf16_t* BT, long ldbt, long b_bs,
                 const float* bias, const float* scale, int ss, int so,
                 const float* resid, void* Out, long o_rs, long o_bs,
                 int M, int N, int K, int batch)
{
  const int waves = (M >> 4) * (N >> 6);
  dim3 grid((waves + 7) / 8, batch);
  k_wmma_gemm<EPI><<<grid, dim3(256), 0, st>>>(
      A, lda, a_bs, BT, ldbt, b_bs,
      bias, scale, ss, so, resid, Out, o_rs, o_bs, M, N, K);
}

static void cvt_t(hipStream_t st, const float* in, bf16_t* out, int R, int C, int batch) {
  const long n = (long)R * C * batch;
  int blocks = (int)((n + 255) / 256); if (blocks > 16384) blocks = 16384;
  k_cvt_t_bf16<<<blocks, 256, 0, st>>>(in, out, R, C, n);
}

extern "C" void kernel_launch(void* const* d_in, const int* in_sizes, int n_in,
                              void* d_out, int out_size, void* d_ws, size_t ws_size,
                              hipStream_t stream)
{
  const float* x        = (const float*)d_in[0];
  const float* xf       = (const float*)d_in[1];
  const float* emb      = (const float*)d_in[2];
  const float* src_mask = (const float*)d_in[3];
  const float* norm_g   = (const float*)d_in[4];
  const float* norm_b   = (const float*)d_in[5];
  const float* tnorm_g  = (const float*)d_in[6];
  const float* tnorm_b  = (const float*)d_in[7];
  const float* m_pos    = (const float*)d_in[8];
  const float* m_gate   = (const float*)d_in[9];
  const float* m_fc1w   = (const float*)d_in[10];
  const float* m_fc1b   = (const float*)d_in[11];
  const float* m_fc2w   = (const float*)d_in[12];
  const float* m_fc2b   = (const float*)d_in[13];
  const float* m_projw  = (const float*)d_in[14];
  const float* m_projb  = (const float*)d_in[15];
  const float* t_pos    = (const float*)d_in[16];
  const float* t_gate   = (const float*)d_in[17];
  const float* t_fc1w   = (const float*)d_in[18];
  const float* t_fc1b   = (const float*)d_in[19];
  const float* t_fc2w   = (const float*)d_in[20];
  const float* t_fc2b   = (const float*)d_in[21];
  const float* t_projw  = (const float*)d_in[22];
  const float* t_projb  = (const float*)d_in[23];
  const float* body_w   = (const float*)d_in[24];
  const float* sb_embw  = (const float*)d_in[25];
  const float* sb_embb  = (const float*)d_in[26];
  const float* sb_ng    = (const float*)d_in[27];
  const float* sb_nb    = (const float*)d_in[28];
  const float* sb_outw  = (const float*)d_in[29];
  const float* sb_outb  = (const float*)d_in[30];
  const int*   cond     = (const int*)d_in[31];
  float* out            = (float*)d_out;

  char* ws = (char*)d_ws;
  size_t off = 0;
  auto alloc = [&](size_t bytes) { size_t o = off; off += (bytes + 255) & ~(size_t)255; return o; };

  // persistent: bf16 weights, transposed to [N,K] so B-fragments load contiguously
  bf16_t* wt1t = (bf16_t*)(ws + alloc((size_t)cE * cPt * 4 * cPt * 2));   // 8x[1024,256]
  bf16_t* wt2t = (bf16_t*)(ws + alloc((size_t)cE * 4 * cPt * cPt * 2));   // 8x[256,1024]
  bf16_t* wpt  = (bf16_t*)(ws + alloc((size_t)cPt * 2 * cL * 2));         // [128,256]
  bf16_t* wt1m = (bf16_t*)(ws + alloc((size_t)cE * cL * 4 * cL * 2));     // 8x[256,64]
  bf16_t* wt2m = (bf16_t*)(ws + alloc((size_t)cE * 4 * cL * cL * 2));     // 8x[64,256]
  bf16_t* wpm  = (bf16_t*)(ws + alloc((size_t)cL * 4 * cL * 2));          // [256,64]
  bf16_t* wemb = (bf16_t*)(ws + alloc((size_t)cTEMB * 2 * cD * 2));       // [1024,2048]
  bf16_t* wout = (bf16_t*)(ws + alloc((size_t)cD * cD * 2));              // [512,512]
  // persistent: f32 activations
  float* text_feat   = (float*)(ws + alloc((size_t)NTOK_T * 128 * 4));
  float* motion_feat = (float*)(ws + alloc((size_t)NTOK_M * 256 * 4));
  float* body_feat   = (float*)(ws + alloc((size_t)cB * cT * cD * 4));
  float* y_attn      = (float*)(ws + alloc((size_t)cB * cT * cD * 4));    // [B,H,T,64]
  float* eo          = (float*)(ws + alloc((size_t)cB * 1024 * 4));
  // reusable arena for phase-local scratch
  const size_t arena = off;
  size_t p = arena;
  auto palloc = [&](size_t bytes) { size_t o = p; p += (bytes + 255) & ~(size_t)255; return o; };

  // ---- phase 0: weight conversion fp32 -> bf16 with transpose ----
  cvt_t(stream, t_fc1w, wt1t, cPt, 4 * cPt, cE);        // [256,1024] -> [1024,256]
  cvt_t(stream, t_fc2w, wt2t, 4 * cPt, cPt, cE);        // [1024,256] -> [256,1024]
  cvt_t(stream, t_projw, wpt, cPt, 2 * cL, 1);          // [256,128]  -> [128,256]
  cvt_t(stream, m_fc1w, wt1m, cL, 4 * cL, cE);          // [64,256]   -> [256,64]
  cvt_t(stream, m_fc2w, wt2m, 4 * cL, cL, cE);          // [256,64]   -> [64,256]
  cvt_t(stream, m_projw, wpm, cL, 4 * cL, 1);           // [64,256]   -> [256,64]
  cvt_t(stream, sb_embw, wemb, cTEMB, 2 * cD, 1);       // [2048,1024]-> [1024,2048]
  cvt_t(stream, sb_outw, wout, cD, cD, 1);              // [512,512]  -> [512,512]

  // ---- phase 1: text + motion MoE ----
  p = arena;
  bf16_t* a_t     = (bf16_t*)(ws + palloc((size_t)NTOK_T * cPt * 2));
  float*  gates_t = (float*)(ws + palloc((size_t)NTOK_T * cE * 4));
  bf16_t* h_t     = (bf16_t*)(ws + palloc((size_t)NTOK_T * 1024 * 2));
  float*  y_t     = (float*)(ws + palloc((size_t)NTOK_T * cPt * 4));
  bf16_t* yg_t    = (bf16_t*)(ws + palloc((size_t)NTOK_T * cPt * 2));
  bf16_t* a_m     = (bf16_t*)(ws + palloc((size_t)NTOK_M * cL * 2));
  float*  gates_m = (float*)(ws + palloc((size_t)NTOK_M * cE * 4));
  bf16_t* h_m     = (bf16_t*)(ws + palloc((size_t)NTOK_M * 256 * 2));
  float*  y_m     = (float*)(ws + palloc((size_t)NTOK_M * cL * 4));
  bf16_t* yg_m    = (bf16_t*)(ws + palloc((size_t)NTOK_M * cL * 2));

  hipMemsetAsync(y_t, 0, (size_t)NTOK_T * cPt * 4, stream);
  hipMemsetAsync(y_m, 0, (size_t)NTOK_M * cL * 4, stream);

  k_text_prep<<<NTOK_T / 8, 256, 0, stream>>>(xf, tnorm_g, tnorm_b, t_pos, t_gate, a_t, gates_t);
  k_motion_prep<<<NTOK_M / 8, 256, 0, stream>>>(x, norm_g, norm_b, m_pos, m_gate, a_m, gates_m);

  for (int e = 0; e < cE; ++e) {
    // text expert e: fc1 (4928x256 @ 256x1024) -> gelu bf16; fc2 -> gated accumulate
    gemm<EPI_GELU_BF16>(stream, a_t, cPt, 0, wt1t + (size_t)e * cPt * 1024, cPt, 0,
                        t_fc1b + (size_t)e * 1024, nullptr, 0, 0, nullptr,
                        h_t, 1024, 0, NTOK_T, 1024, cPt, 1);
    gemm<EPI_SCALED_ADD>(stream, h_t, 1024, 0, wt2t + (size_t)e * 1024 * cPt, 1024, 0,
                         t_fc2b + (size_t)e * cPt, gates_t, cE, e, nullptr,
                         y_t, cPt, 0, NTOK_T, cPt, 1024, 1);
    // motion expert e: fc1 (131072x64 @ 64x256); fc2 (131072x256 @ 256x64)
    gemm<EPI_GELU_BF16>(stream, a_m, cL, 0, wt1m + (size_t)e * cL * 256, cL, 0,
                        m_fc1b + (size_t)e * 256, nullptr, 0, 0, nullptr,
                        h_m, 256, 0, NTOK_M, 256, cL, 1);
    gemm<EPI_SCALED_ADD>(stream, h_m, 256, 0, wt2m + (size_t)e * 256 * cL, 256, 0,
                         m_fc2b + (size_t)e * cL, gates_m, cE, e, nullptr,
                         y_m, cL, 0, NTOK_M, cL, 256, 1);
  }

  // gelu(y) @ proj + bias
  k_gelu_cvt_bf16<<<16384, 256, 0, stream>>>(y_t, yg_t, (long)NTOK_T * cPt);
  gemm<EPI_F32>(stream, yg_t, cPt, 0, wpt, cPt, 0, t_projb, nullptr, 0, 0, nullptr,
                text_feat, 128, 0, NTOK_T, 128, cPt, 1);
  k_gelu_cvt_bf16<<<16384, 256, 0, stream>>>(y_m, yg_m, (long)NTOK_M * cL);
  gemm<EPI_F32>(stream, yg_m, cL, 0, wpm, cL, 0, m_projb, nullptr, 0, 0, nullptr,
                motion_feat, 256, 0, NTOK_M, 256, cL, 1);

  // ---- phase 2: body mixing + attention ----
  p = arena;  // phase-1 scratch now dead
  bf16_t* keyPT = (bf16_t*)(ws + palloc((size_t)cB * cH * 64 * cNPAD * 2)); // [bh][d][n]
  bf16_t* valPT = (bf16_t*)(ws + palloc((size_t)cB * cH * 64 * cNPAD * 2)); // [bh][l][n]
  bf16_t* attnT = (bf16_t*)(ws + palloc((size_t)cB * cH * 64 * 64 * 2));    // [bh][l][d]
  bf16_t* qbf   = (bf16_t*)(ws + palloc((size_t)cB * cH * cT * 64 * 2));    // [bh][t][d]

  k_bodymix<<<(int)(((long)cB * cT * cH * cL + 255) / 256), 256, 0, stream>>>(
      motion_feat, body_w, body_feat);
  k_build_key<<<(cB * cH * 2) / 8, 256, 0, stream>>>(text_feat, motion_feat, src_mask, cond, keyPT);
  k_build_value<<<(int)(((long)cB * cH * 64 * cNPAD + 255) / 256), 256, 0, stream>>>(
      text_feat, motion_feat, src_mask, cond, valPT);
  // attn[b,h] = K^T (64 x 352) @ V (352 x 64); D stored transposed -> attnT [l][d]
  gemm<EPI_BF16_T>(stream, keyPT, cNPAD, 64L * cNPAD, valPT, cNPAD, 64L * cNPAD,
                   nullptr, nullptr, 0, 0, nullptr, attnT, 64, 64L * 64,
                   64, 64, cNPAD, cB * cH);
  k_query<<<(cB * cH * cT) / 8, 256, 0, stream>>>(motion_feat, qbf);
  // y[b,h] = Q (256 x 64) @ attn (64 x 64); BT = attnT
  gemm<EPI_F32>(stream, qbf, 64, (long)cT * 64, attnT, 64, 64L * 64,
                nullptr, nullptr, 0, 0, nullptr, y_attn, 64, (long)cT * 64,
                cT, 64, 64, cB * cH);

  // ---- phase 3: stylization ----
  p = arena;
  bf16_t* embs = (bf16_t*)(ws + palloc((size_t)cB * cTEMB * 2));
  bf16_t* hsil = (bf16_t*)(ws + palloc((size_t)cB * cT * cD * 2));

  k_silu_cvt_bf16<<<512, 256, 0, stream>>>(emb, embs, (long)cB * cTEMB);
  gemm<EPI_F32>(stream, embs, cTEMB, 0, wemb, cTEMB, 0, sb_embb, nullptr, 0, 0, nullptr,
                eo, 2 * cD, 0, cB, 2 * cD, cTEMB, 1);
  k_styl<<<(cB * cT) / 8, 256, 0, stream>>>(body_feat, y_attn, eo, sb_ng, sb_nb, hsil);
  // out = x + silu(h) @ sb_outw + sb_outb
  gemm<EPI_RESID>(stream, hsil, cD, 0, wout, cD, 0, sb_outb, nullptr, 0, 0, x,
                  out, cD, 0, cB * cT, cD, cD, 1);

  (void)in_sizes; (void)n_in; (void)out_size; (void)ws_size;
}
